// CausalSelfAttention_33011118637692
// MI455X (gfx1250) — matrix operand order repaired and checked by reading, not compiled, not run on hardware
//
#include <hip/hip_runtime.h>
#include <cstdint>

typedef __attribute__((ext_vector_type(16))) __bf16 v16bf;
typedef __attribute__((ext_vector_type(8)))  __bf16 v8bf;
typedef __attribute__((ext_vector_type(8)))  float  v8f;
typedef __attribute__((ext_vector_type(4)))  int    v4i;

typedef v4i __attribute__((address_space(1)))* gv4i_p;   // global
typedef v4i __attribute__((address_space(3)))* lv4i_p;   // LDS

union BF16x16 { v16bf v; v8bf h[2]; };

constexpr int BB  = 2;
constexpr int TT  = 2048;
constexpr int DD  = 1024;
constexpr int HH  = 16;
constexpr int HDD = 64;
constexpr float SCALE  = 0.125f;     // 1/sqrt(64)
constexpr float NEGBIG = -3.0e38f;

#if __has_builtin(__builtin_amdgcn_global_load_async_to_lds_b128) && \
    __has_builtin(__builtin_amdgcn_s_wait_asynccnt)
#define ASYNC_LDS 1
#else
#define ASYNC_LDS 0
#endif

static __device__ __forceinline__ v8f wmma_bf16(v16bf a, v16bf b, v8f c) {
  return __builtin_amdgcn_wmma_f32_16x16x32_bf16(false, a, false, b, (short)0, c, false, false);
}

// ---------------------------------------------------------------- utilities
__global__ void f2bf_kernel(const float* __restrict__ in, __bf16* __restrict__ out, int n) {
  int i = blockIdx.x * blockDim.x + threadIdx.x;
  if (i < n) out[i] = (__bf16)in[i];
}

// Wt[n*K + k] = (bf16) W[k*N + n]   (W is K x N row-major)
__global__ void transpose_f2bf_kernel(const float* __restrict__ W, __bf16* __restrict__ Wt,
                                      int K, int N) {
  int i = blockIdx.x * blockDim.x + threadIdx.x;
  if (i >= K * N) return;
  int k = i / N, n = i % N;
  Wt[(size_t)n * K + k] = (__bf16)W[(size_t)k * N + n];
}

// ---------------------------------------------------------------- GEMM
// C(MxN) = A(MxK bf16 row-major) * B(KxN), Bt = B^T (NxK bf16).
// Block = 128 threads = 4 waves; block tile 64x64 (waves stacked in M).
// B tile (64 N-rows x 32 K-halves) is double-buffered in LDS and staged with
// async global->LDS copies so WMMAs on buffer b overlap the fill of buffer b^1.
__global__ __launch_bounds__(128) void gemm_bf16_wmma(
    const __bf16* __restrict__ A, const __bf16* __restrict__ Bt,
    float* __restrict__ Cf, __bf16* __restrict__ Cb,
    int M, int N, int K, int store_bf16) {
  __shared__ __bf16 bs[2][64 * 32];   // 2 x 4 KB

  int ntn  = N >> 6;
  int tm   = blockIdx.x / ntn;        // 64-row M tile
  int tn   = blockIdx.x % ntn;        // 64-col N tile
  int wid  = threadIdx.x >> 5;
  int lane = threadIdx.x & 31;
  int g    = lane >> 4;               // lane half
  int ln   = lane & 15;
  int tid  = threadIdx.x;

  // Each thread stages 16 halves (two b128) of the 2048-half B tile.
  // Flat half index f = n*32 + kk  ->  n = f>>5, kk = f&31 (8-half chunks stay in-row).
  int f0  = tid * 16;
  int sn  = f0 >> 5;
  int skk = f0 & 31;

  auto stage = [&](int buf, int k0) {
    const __bf16* src = Bt + (size_t)(tn * 64 + sn) * K + k0 + skk;
    __bf16* dst = &bs[buf][f0];
#if ASYNC_LDS
    __builtin_amdgcn_global_load_async_to_lds_b128((gv4i_p)src,       (lv4i_p)dst,       0, 0);
    __builtin_amdgcn_global_load_async_to_lds_b128((gv4i_p)(src + 8), (lv4i_p)(dst + 8), 0, 0);
#else
    *(v8bf*)dst       = *(const v8bf*)src;
    *(v8bf*)(dst + 8) = *(const v8bf*)(src + 8);
#endif
  };

  const __bf16* arow = A + (size_t)(tm * 64 + wid * 16 + ln) * K;
  v8f acc[4] = {};

  stage(0, 0);
#if ASYNC_LDS
  __builtin_amdgcn_s_wait_asynccnt(0);
#endif
  __syncthreads();

  int nk = K >> 5;
  for (int kt = 0; kt < nk; ++kt) {
    int buf = kt & 1;
    int k0  = kt << 5;
    if (kt + 1 < nk) stage(buf ^ 1, k0 + 32);   // fill next buffer in flight

    // A-frag (16x32): half g -> K = k0+16g..+15 contiguous
    BF16x16 a;
    const __bf16* pa = arow + k0 + 16 * g;
    a.h[0] = *(const v8bf*)pa;
    a.h[1] = *(const v8bf*)(pa + 8);

#pragma unroll
    for (int j = 0; j < 4; ++j) {
      // B-frag (32x16): lane col = ln, K contiguous 16 halves at offset 16g
      BF16x16 b;
      const __bf16* pb = &bs[buf][(j * 16 + ln) * 32 + 16 * g];
      b.h[0] = *(const v8bf*)pb;
      b.h[1] = *(const v8bf*)(pb + 8);
      acc[j] = wmma_bf16(a.v, b.v, acc[j]);
    }

#if ASYNC_LDS
    __builtin_amdgcn_s_wait_asynccnt(0);
#endif
    __syncthreads();   // next buffer filled; current buffer free for reuse
  }

  // C-frag: VGPR r -> row 8g+r, lane ln -> col
#pragma unroll
  for (int j = 0; j < 4; ++j)
#pragma unroll
    for (int r = 0; r < 8; ++r) {
      int row = tm * 64 + wid * 16 + 8 * g + r;
      int col = tn * 64 + j * 16 + ln;
      if (store_bf16) Cb[(size_t)row * N + col] = (__bf16)acc[j][r];
      else            Cf[(size_t)row * N + col] = acc[j][r];
    }
}

// ---------------------------------------------------------------- RoPE + split
// qkv: (B*T, 3*D) bf16; outputs: qr,kr (b,h,t,hd) rotated; vt (b,h,hd,t) transposed.
__global__ void rope_split_kernel(const __bf16* __restrict__ qkv,
                                  const float* __restrict__ fc, const float* __restrict__ fs,
                                  __bf16* __restrict__ qr, __bf16* __restrict__ kr,
                                  __bf16* __restrict__ vt) {
  int idx = blockIdx.x * blockDim.x + threadIdx.x;
  if (idx >= BB * TT * HH * (HDD / 2)) return;
  int i = idx & 31;                 // rotation pair index (HD/2 = 32)
  int h = (idx >> 5) & (HH - 1);
  int t = (idx >> 9) & (TT - 1);
  int b = idx >> 20;

  size_t row = (size_t)(b * TT + t) * (3 * DD);
  int qc = h * HDD + 2 * i;
  float c = fc[t * 32 + i], s = fs[t * 32 + i];
  float qe = (float)qkv[row + qc],           qo = (float)qkv[row + qc + 1];
  float ke = (float)qkv[row + DD + qc],      ko = (float)qkv[row + DD + qc + 1];
  float v0 = (float)qkv[row + 2 * DD + qc],  v1 = (float)qkv[row + 2 * DD + qc + 1];

  size_t ob = ((size_t)(b * HH + h) * TT + t) * HDD;
  qr[ob + 2 * i]     = (__bf16)(qe * c - qo * s);
  qr[ob + 2 * i + 1] = (__bf16)(qe * s + qo * c);
  kr[ob + 2 * i]     = (__bf16)(ke * c - ko * s);
  kr[ob + 2 * i + 1] = (__bf16)(ke * s + ko * c);

  size_t vbase = (size_t)(b * HH + h) * HDD * TT;
  vt[vbase + (size_t)(2 * i) * TT + t]     = (__bf16)v0;
  vt[vbase + (size_t)(2 * i + 1) * TT + t] = (__bf16)v1;
}

// ---------------------------------------------------------------- flash attention
// One wave per 16-row Q tile per (b,h). Key tiles of 32, online softmax.
__global__ __launch_bounds__(32) void flash_attn_wmma(
    const __bf16* __restrict__ qr, const __bf16* __restrict__ kr,
    const __bf16* __restrict__ vt, __bf16* __restrict__ ao) {
  __shared__ __bf16 lds[16][48];          // 16x32 P tile, pitch 48 halves (96 B, 16B-aligned)

  int tile = blockIdx.x;
  int qt = tile & (TT / 16 - 1);
  int bh = tile / (TT / 16);
  int h  = bh & (HH - 1);
  int b  = bh / HH;
  int q0 = qt * 16;
  int lane = threadIdx.x & 31;
  int g  = lane >> 4;
  int ln = lane & 15;

  const __bf16* qb = qr + (size_t)bh * TT * HDD;
  const __bf16* kb = kr + (size_t)bh * TT * HDD;
  const __bf16* vb = vt + (size_t)bh * HDD * TT;

  // Q A-frags: rows q0..q0+15, hd steps {0,32}
  BF16x16 aq[2];
#pragma unroll
  for (int f = 0; f < 2; ++f) {
    const __bf16* p = qb + (size_t)(q0 + ln) * HDD + f * 32 + 16 * g;
    aq[f].h[0] = *(const v8bf*)p;
    aq[f].h[1] = *(const v8bf*)(p + 8);
  }

  float m_r[8], l_r[8];
  v8f o[4] = {};
#pragma unroll
  for (int r = 0; r < 8; ++r) { m_r[r] = NEGBIG; l_r[r] = 0.0f; }

  int nkt = (q0 + 47) >> 5;               // covers keys 0..q0+15
  for (int kt = 0; kt < nkt; ++kt) {
    int j0 = kt * 32;
    float sv[2][8];
#pragma unroll
    for (int ns = 0; ns < 2; ++ns) {
      v8f s = {};
#pragma unroll
      for (int f = 0; f < 2; ++f) {
        // K^T B-frag: lane col = key j0+ns*16+ln, K = hd f*32+16g .. +15 contiguous
        BF16x16 bk;
        const __bf16* p = kb + (size_t)(j0 + ns * 16 + ln) * HDD + f * 32 + 16 * g;
        bk.h[0] = *(const v8bf*)p;
        bk.h[1] = *(const v8bf*)(p + 8);
        s = wmma_bf16(aq[f].v, bk.v, s);
      }
      int col = j0 + ns * 16 + ln;
#pragma unroll
      for (int r = 0; r < 8; ++r) {
        int row = q0 + 8 * g + r;
        float v = s[r] * SCALE;
        sv[ns][r] = (col > row) ? NEGBIG : v;   // causal mask, no EXEC change
      }
    }
    // row max across 16 lanes of each half (rows 8g..8g+7 in reg index r)
    float mt[8];
#pragma unroll
    for (int r = 0; r < 8; ++r) mt[r] = fmaxf(sv[0][r], sv[1][r]);
#pragma unroll
    for (int off = 1; off < 16; off <<= 1)
#pragma unroll
      for (int r = 0; r < 8; ++r) mt[r] = fmaxf(mt[r], __shfl_xor(mt[r], off, 32));

    float al[8];
#pragma unroll
    for (int r = 0; r < 8; ++r) {
      float mn = fmaxf(m_r[r], mt[r]);
      al[r] = __expf(m_r[r] - mn);
      m_r[r] = mn;
    }
    // P = exp(S - m); stage to LDS for C-frag -> A-frag transpose
#pragma unroll
    for (int ns = 0; ns < 2; ++ns)
#pragma unroll
      for (int r = 0; r < 8; ++r) {
        float p = __expf(sv[ns][r] - m_r[r]);
        sv[ns][r] = p;
        lds[8 * g + r][ns * 16 + ln] = (__bf16)p;
      }
    float rs[8];
#pragma unroll
    for (int r = 0; r < 8; ++r) rs[r] = sv[0][r] + sv[1][r];
#pragma unroll
    for (int off = 1; off < 16; off <<= 1)
#pragma unroll
      for (int r = 0; r < 8; ++r) rs[r] += __shfl_xor(rs[r], off, 32);
#pragma unroll
    for (int r = 0; r < 8; ++r) l_r[r] = l_r[r] * al[r] + rs[r];
#pragma unroll
    for (int nd = 0; nd < 4; ++nd)
#pragma unroll
      for (int r = 0; r < 8; ++r) o[nd][r] *= al[r];

    __syncthreads();
    // P A-frag: lane row = ln, keys 16g..16g+15 contiguous
    BF16x16 ap;
    {
      const __bf16* lp = &lds[ln][0];
      ap.h[0] = *(const v8bf*)(lp + 16 * g);
      ap.h[1] = *(const v8bf*)(lp + 16 * g + 8);
    }
    // O += P @ V : V^T stored (hd, t) -> B-frag contiguous keys
#pragma unroll
    for (int nd = 0; nd < 4; ++nd) {
      BF16x16 bv;
      const __bf16* p = vb + (size_t)(nd * 16 + ln) * TT + j0 + 16 * g;
      bv.h[0] = *(const v8bf*)p;
      bv.h[1] = *(const v8bf*)(p + 8);
      o[nd] = wmma_bf16(ap.v, bv.v, o[nd]);
    }
    __syncthreads();
  }

  // write attention output directly into (b, t, D) bf16 layout for the final GEMM
#pragma unroll
  for (int nd = 0; nd < 4; ++nd)
#pragma unroll
    for (int r = 0; r < 8; ++r) {
      int t   = q0 + 8 * g + r;
      int col = h * HDD + nd * 16 + ln;
      ao[(size_t)(b * TT + t) * DD + col] = (__bf16)(o[nd][r] / l_r[r]);
    }
}

// ---------------------------------------------------------------- launch
extern "C" void kernel_launch(void* const* d_in, const int* in_sizes, int n_in,
                              void* d_out, int out_size, void* d_ws, size_t ws_size,
                              hipStream_t stream) {
  const float* x     = (const float*)d_in[0];
  const float* w_qkv = (const float*)d_in[1];
  const float* w_out = (const float*)d_in[2];
  const float* fcos  = (const float*)d_in[3];
  const float* fsin  = (const float*)d_in[4];
  (void)in_sizes; (void)n_in; (void)out_size; (void)ws_size;

  char* ws = (char*)d_ws;
  size_t off = 0;
  auto carve = [&](size_t bytes) -> void* {
    void* p = ws + off;
    off += (bytes + 255) & ~(size_t)255;
    return p;
  };
  const size_t ntok = (size_t)BB * TT;                     // 4096
  __bf16* xb   = (__bf16*)carve(ntok * DD * 2);            // x bf16
  __bf16* wqt  = (__bf16*)carve((size_t)3 * DD * DD * 2);  // w_qkv^T bf16
  __bf16* wot  = (__bf16*)carve((size_t)DD * DD * 2);      // w_out^T bf16
  __bf16* qkvb = (__bf16*)carve(ntok * 3 * DD * 2);        // qkv bf16
  __bf16* qrb  = (__bf16*)carve(ntok * DD * 2);            // q roped (b,h,t,hd)
  __bf16* krb  = (__bf16*)carve(ntok * DD * 2);            // k roped (b,h,t,hd)
  __bf16* vtb  = (__bf16*)carve(ntok * DD * 2);            // v transposed (b,h,hd,t)
  __bf16* aob  = (__bf16*)carve(ntok * DD * 2);            // attn out (b,t,D)

  {
    int n = (int)(ntok * DD);
    f2bf_kernel<<<(n + 255) / 256, 256, 0, stream>>>(x, xb, n);
  }
  {
    int n = DD * 3 * DD;
    transpose_f2bf_kernel<<<(n + 255) / 256, 256, 0, stream>>>(w_qkv, wqt, DD, 3 * DD);
  }
  {
    int n = DD * DD;
    transpose_f2bf_kernel<<<(n + 255) / 256, 256, 0, stream>>>(w_out, wot, DD, DD);
  }
  {
    int M = (int)ntok, N = 3 * DD, K = DD;
    int nblk = (M / 64) * (N / 64);
    gemm_bf16_wmma<<<nblk, 128, 0, stream>>>(xb, wqt, nullptr, qkvb, M, N, K, 1);
  }
  {
    int n = BB * TT * HH * (HDD / 2);
    rope_split_kernel<<<(n + 255) / 256, 256, 0, stream>>>(qkvb, fcos, fsin, qrb, krb, vtb);
  }
  {
    int nblk = BB * HH * (TT / 16);
    flash_attn_wmma<<<nblk, 32, 0, stream>>>(qrb, krb, vtb, aob);
  }
  {
    int M = (int)ntok, N = DD, K = DD;
    int nblk = (M / 64) * (N / 64);
    gemm_bf16_wmma<<<nblk, 128, 0, stream>>>(aob, wot, (float*)d_out, nullptr, M, N, K, 0);
  }
}
